// MultiHeadSelfAttention_85804856639663
// MI455X (gfx1250) — compile-verified, matrix operand order checked by reading, not run on hardware
//
#include <hip/hip_runtime.h>
#include <stdint.h>

// ---------------- CDNA5 / gfx1250 WMMA types ----------------
typedef __attribute__((ext_vector_type(16))) __bf16 v16bf;
typedef __attribute__((ext_vector_type(8)))  float  v8f;

#define D_MODEL 2048
#define SEQ     2048
#define BATCH   2
#define NHEAD   16
#define DKH     128
#define MTOT    (BATCH*SEQ)   // 4096

union FragBF { uint4 q[2]; v16bf v; };

__device__ inline unsigned short f2bf(float f) {
  unsigned u = __builtin_bit_cast(unsigned, f);
  u += 0x7FFFu + ((u >> 16) & 1u);      // round-to-nearest-even
  return (unsigned short)(u >> 16);
}

__device__ inline v8f v8zero() {
  v8f z;
#pragma unroll
  for (int i = 0; i < 8; ++i) z[i] = 0.f;
  return z;
}

// ---- CDNA5 async global->LDS DMA (ASYNCcnt-tracked, no VGPR round trip) ----
__device__ inline void async_copy_b128(const unsigned short* gptr,
                                       const unsigned short* lptr) {
  unsigned lds = (unsigned)(uintptr_t)lptr;   // flat LDS aperture: low 32b = ds offset
  asm volatile("global_load_async_to_lds_b128 %0, %1, off"
               :: "v"(lds), "v"(gptr)
               : "memory");
}
__device__ inline void wait_async0() {
  asm volatile("s_wait_asynccnt 0" ::: "memory");
}

// ---------------- f32 -> bf16 conversion ----------------
__global__ void cvt_f32_bf16(const float* __restrict__ src,
                             unsigned short* __restrict__ dst, int n) {
  for (int i = blockIdx.x * blockDim.x + threadIdx.x; i < n;
       i += gridDim.x * blockDim.x)
    dst[i] = f2bf(src[i]);
}

// ---------------- NT GEMM: C[M,N] = A[M,K] * W[N,K]^T + bias ----------------
// block = 256 threads (8 waves), block tile 128x128, wave = 16x128 strip.
// Double-buffered LDS staged with global_load_async_to_lds_b128: tile k+1 is
// DMA'd while tile k runs on the WMMA pipe.
// fmode < 0: mode = blockIdx.z in {0,1,2} -> Q / K / V-transposed bf16 epilogues
// fmode ==3: f32 row-major output (final projection)
__global__ __launch_bounds__(256)
void gemm_nt_wmma(const unsigned short* __restrict__ A,
                  const unsigned short* __restrict__ Wall,
                  const float* __restrict__ b0, const float* __restrict__ b1,
                  const float* __restrict__ b2,
                  unsigned short* __restrict__ Qo,
                  unsigned short* __restrict__ Ko,
                  unsigned short* __restrict__ Vto,
                  float* __restrict__ Fo, int fmode) {
  const int mode = (fmode < 0) ? (int)blockIdx.z : fmode;
  const unsigned short* W = Wall + (size_t)((mode < 3) ? mode : 0) * D_MODEL * D_MODEL;
  const float* bias = (mode == 0) ? b0 : (mode == 1) ? b1 : b2;

  __shared__ unsigned short xs[2][128][72];   // +8 pad, row = 144B (16B aligned)
  __shared__ unsigned short wsm[2][128][72];  // 2x double buffer, 72KB total

  const int tid  = threadIdx.x;
  const int w    = tid >> 5;
  const int lane = tid & 31;
  const int lr   = lane & 15;            // row (A) / col (B) within 16
  const int kh   = (lane >> 4) << 3;     // K-chunk base: 0 or 8
  const int ml   = (lane >> 4) << 3;     // C/D row offset: 0 or 8

  const int m0 = blockIdx.y * 128;
  const int n0 = blockIdx.x * 128;

  v8f acc[8];
#pragma unroll
  for (int j = 0; j < 8; ++j) acc[j] = v8zero();

  // per-thread chunk coords for cooperative 128x64 tile DMA (16B chunks)
  int crow[4], ccol[4];
#pragma unroll
  for (int it = 0; it < 4; ++it) {
    int c = tid + it * 256;            // 0..1023
    crow[it] = c >> 3;
    ccol[it] = (c & 7) << 3;
  }

  // prologue: DMA first tile into buffer 0
#pragma unroll
  for (int it = 0; it < 4; ++it) {
    async_copy_b128(A + (size_t)(m0 + crow[it]) * D_MODEL + ccol[it],
                    &xs[0][crow[it]][ccol[it]]);
    async_copy_b128(W + (size_t)(n0 + crow[it]) * D_MODEL + ccol[it],
                    &wsm[0][crow[it]][ccol[it]]);
  }
  wait_async0();
  __syncthreads();

  int cur = 0;
  for (int k0 = 0; k0 < D_MODEL; k0 += 64) {
    // DMA next tile into the other buffer while this one computes
    if (k0 + 64 < D_MODEL) {
      const int nxt = cur ^ 1;
#pragma unroll
      for (int it = 0; it < 4; ++it) {
        async_copy_b128(A + (size_t)(m0 + crow[it]) * D_MODEL + k0 + 64 + ccol[it],
                        &xs[nxt][crow[it]][ccol[it]]);
        async_copy_b128(W + (size_t)(n0 + crow[it]) * D_MODEL + k0 + 64 + ccol[it],
                        &wsm[nxt][crow[it]][ccol[it]]);
      }
    }

    const unsigned short (*xsb)[72]  = xs[cur];
    const unsigned short (*wsmb)[72] = wsm[cur];
#pragma unroll
    for (int kk = 0; kk < 64; kk += 32) {
      FragBF a;
      a.q[0] = *(const uint4*)&xsb[w * 16 + lr][kk + kh];
      a.q[1] = *(const uint4*)&xsb[w * 16 + lr][kk + kh + 16];
#pragma unroll
      for (int j = 0; j < 8; ++j) {
        FragBF bfr;
        bfr.q[0] = *(const uint4*)&wsmb[j * 16 + lr][kk + kh];
        bfr.q[1] = *(const uint4*)&wsmb[j * 16 + lr][kk + kh + 16];
        acc[j] = __builtin_amdgcn_wmma_f32_16x16x32_bf16(
            false, a.v, false, bfr.v, (short)0, acc[j], false, false);
      }
    }
    wait_async0();        // next tile landed (overlapped with the WMMAs above)
    __syncthreads();      // all waves done reading cur before it is rewritten
    cur ^= 1;
  }

  // epilogue: bias + scatter
#pragma unroll
  for (int j = 0; j < 8; ++j) {
    const int n_g = n0 + j * 16 + lr;
    const float bval = bias[n_g];
#pragma unroll
    for (int v = 0; v < 8; ++v) {
      const int m_g = m0 + w * 16 + v + ml;
      const float val = acc[j][v] + bval;
      if (mode == 3) {
        Fo[(size_t)m_g * D_MODEL + n_g] = val;
      } else {
        const int b = m_g >> 11, s = m_g & 2047;
        const int h = n_g >> 7,  d = n_g & 127;
        if (mode == 2) {
          Vto[(((size_t)(b * NHEAD + h)) * DKH + d) * SEQ + s] = f2bf(val);
        } else {
          unsigned short* dst = (mode == 0) ? Qo : Ko;
          dst[(((size_t)(b * NHEAD + h)) * SEQ + s) * DKH + d] = f2bf(val);
        }
      }
    }
  }
}

// ---------------- causal flash attention ----------------
// grid = (S/128, H, B), block = 256. Wave w owns q rows [q0+16w, q0+16w+16).
__global__ __launch_bounds__(256)
void attn_flash_wmma(const unsigned short* __restrict__ Q,
                     const unsigned short* __restrict__ K,
                     const unsigned short* __restrict__ Vt,
                     unsigned short* __restrict__ Ctx) {
  __shared__ unsigned short Pb[8][16][136];  // per-wave P tile, row = 272B

  const int tid  = threadIdx.x;
  const int w    = tid >> 5;
  const int lane = tid & 31;
  const int lr   = lane & 15;
  const int kh   = (lane >> 4) << 3;
  const int ml   = (lane >> 4) << 3;

  const int qblk = blockIdx.x, h = blockIdx.y, b = blockIdx.z;
  const int q0 = qblk * 128;
  const size_t hb = (size_t)(b * NHEAD + h);
  const unsigned short* Qh = Q  + hb * SEQ * DKH;
  const unsigned short* Kh = K  + hb * SEQ * DKH;
  const unsigned short* Vh = Vt + hb * DKH * SEQ;

  // Q A-fragments over dk = 128 (4 k-steps of 32), resident for whole kernel
  v16bf qf[4];
  {
    const unsigned short* qr = Qh + (size_t)(q0 + w * 16 + lr) * DKH;
#pragma unroll
    for (int kk = 0; kk < 4; ++kk) {
      FragBF f;
      f.q[0] = *(const uint4*)(qr + kk * 32 + kh);
      f.q[1] = *(const uint4*)(qr + kk * 32 + kh + 16);
      qf[kk] = f.v;
    }
  }

  v8f acc[8];
#pragma unroll
  for (int jd = 0; jd < 8; ++jd) acc[jd] = v8zero();
  float mrun[8], lrun[8];
#pragma unroll
  for (int v = 0; v < 8; ++v) { mrun[v] = -1e30f; lrun[v] = 0.f; }

  const float scale = 0.08838834764831845f;  // 1/sqrt(128)

  for (int kb = 0; kb <= qblk; ++kb) {
    // scores S (16 x 128) = Q . K^T
    v8f sc[8];
#pragma unroll
    for (int j = 0; j < 8; ++j) {
      const unsigned short* kr = Kh + (size_t)(kb * 128 + j * 16 + lr) * DKH;
      v8f s = v8zero();
#pragma unroll
      for (int kk = 0; kk < 4; ++kk) {
        FragBF f;
        f.q[0] = *(const uint4*)(kr + kk * 32 + kh);
        f.q[1] = *(const uint4*)(kr + kk * 32 + kh + 16);
        s = __builtin_amdgcn_wmma_f32_16x16x32_bf16(
            false, qf[kk], false, f.v, (short)0, s, false, false);
      }
      sc[j] = s;
    }
    // scale + causal mask
#pragma unroll
    for (int j = 0; j < 8; ++j) {
      const int col = kb * 128 + j * 16 + lr;
#pragma unroll
      for (int v = 0; v < 8; ++v) {
        const int row = q0 + w * 16 + v + ml;
        const float x = sc[j][v] * scale;
        sc[j][v] = (col > row) ? -1e30f : x;
      }
    }
    // online softmax (rows live in half-waves -> width-16 xor reductions)
    float alpha[8];
#pragma unroll
    for (int v = 0; v < 8; ++v) {
      float t = sc[0][v];
#pragma unroll
      for (int j = 1; j < 8; ++j) t = fmaxf(t, sc[j][v]);
#pragma unroll
      for (int off = 1; off < 16; off <<= 1) t = fmaxf(t, __shfl_xor(t, off, 16));
      const float mnew = fmaxf(mrun[v], t);
      alpha[v] = __expf(mrun[v] - mnew);
      mrun[v] = mnew;
      float rs = 0.f;
#pragma unroll
      for (int j = 0; j < 8; ++j) {
        const float p = __expf(sc[j][v] - mnew);
        sc[j][v] = p;
        rs += p;
      }
#pragma unroll
      for (int off = 1; off < 16; off <<= 1) rs += __shfl_xor(rs, off, 16);
      lrun[v] = lrun[v] * alpha[v] + rs;
    }
#pragma unroll
    for (int jd = 0; jd < 8; ++jd)
#pragma unroll
      for (int v = 0; v < 8; ++v) acc[jd][v] *= alpha[v];

    // C-layout P -> LDS -> A-fragment layout
#pragma unroll
    for (int j = 0; j < 8; ++j)
#pragma unroll
      for (int v = 0; v < 8; ++v)
        Pb[w][v + ml][j * 16 + lr] = f2bf(sc[j][v]);
    __syncthreads();

    v16bf pf[4];
#pragma unroll
    for (int kk = 0; kk < 4; ++kk) {
      FragBF f;
      f.q[0] = *(const uint4*)&Pb[w][lr][kk * 32 + kh];
      f.q[1] = *(const uint4*)&Pb[w][lr][kk * 32 + kh + 16];
      pf[kk] = f.v;
    }
    // ctx += P @ V  (V transposed: rows are d, contiguous over keys)
#pragma unroll
    for (int jd = 0; jd < 8; ++jd) {
      const unsigned short* vr = Vh + (size_t)(jd * 16 + lr) * SEQ + kb * 128;
      v8f a = acc[jd];
#pragma unroll
      for (int kk = 0; kk < 4; ++kk) {
        FragBF f;
        f.q[0] = *(const uint4*)(vr + kk * 32 + kh);
        f.q[1] = *(const uint4*)(vr + kk * 32 + kh + 16);
        a = __builtin_amdgcn_wmma_f32_16x16x32_bf16(
            false, pf[kk], false, f.v, (short)0, a, false, false);
      }
      acc[jd] = a;
    }
    __syncthreads();
  }

  // normalize + store ctx (bf16, merged heads [B*S, D])
#pragma unroll
  for (int v = 0; v < 8; ++v) {
    const float inv = 1.f / lrun[v];
#pragma unroll
    for (int jd = 0; jd < 8; ++jd) acc[jd][v] *= inv;
  }
#pragma unroll
  for (int jd = 0; jd < 8; ++jd) {
    const int col = h * DKH + jd * 16 + lr;
#pragma unroll
    for (int v = 0; v < 8; ++v) {
      const int row = b * SEQ + q0 + w * 16 + v + ml;
      Ctx[(size_t)row * D_MODEL + col] = f2bf(acc[jd][v]);
    }
  }
}

// ---------------- host launcher ----------------
extern "C" void kernel_launch(void* const* d_in, const int* in_sizes, int n_in,
                              void* d_out, int out_size, void* d_ws, size_t ws_size,
                              hipStream_t stream) {
  (void)in_sizes; (void)n_in; (void)out_size; (void)ws_size;
  const float* x  = (const float*)d_in[0];
  const float* Wq = (const float*)d_in[1];
  const float* bq = (const float*)d_in[2];
  const float* Wk = (const float*)d_in[3];
  const float* bk = (const float*)d_in[4];
  const float* Wv = (const float*)d_in[5];
  const float* bv = (const float*)d_in[6];
  const float* Wo = (const float*)d_in[7];
  const float* bo = (const float*)d_in[8];
  float* out = (float*)d_out;

  const size_t DD = (size_t)D_MODEL * D_MODEL;  // 4,194,304
  const size_t MD = (size_t)MTOT * D_MODEL;     // 8,388,608

  unsigned short* xb  = (unsigned short*)d_ws;   // [4096,2048] bf16
  unsigned short* Wb  = xb + MD;                 // Wq,Wk,Wv,Wo bf16 stacked
  unsigned short* Qw  = Wb + 4 * DD;             // [B,H,S,128]
  unsigned short* Kw  = Qw + MD;                 // [B,H,S,128]
  unsigned short* Vtw = Kw + MD;                 // [B,H,128,S]
  unsigned short* Cw  = Vtw + MD;                // [4096,2048]

  cvt_f32_bf16<<<4096, 256, 0, stream>>>(x,  xb,          (int)MD);
  cvt_f32_bf16<<<4096, 256, 0, stream>>>(Wq, Wb + 0 * DD, (int)DD);
  cvt_f32_bf16<<<4096, 256, 0, stream>>>(Wk, Wb + 1 * DD, (int)DD);
  cvt_f32_bf16<<<4096, 256, 0, stream>>>(Wv, Wb + 2 * DD, (int)DD);
  cvt_f32_bf16<<<4096, 256, 0, stream>>>(Wo, Wb + 3 * DD, (int)DD);

  // fused Q/K/V projections (gridDim.z selects head-split epilogue)
  gemm_nt_wmma<<<dim3(D_MODEL / 128, MTOT / 128, 3), 256, 0, stream>>>(
      xb, Wb, bq, bk, bv, Qw, Kw, Vtw, nullptr, -1);

  // causal flash attention
  attn_flash_wmma<<<dim3(SEQ / 128, NHEAD, BATCH), 256, 0, stream>>>(
      Qw, Kw, Vtw, Cw);

  // output projection -> f32
  gemm_nt_wmma<<<dim3(D_MODEL / 128, MTOT / 128, 1), 256, 0, stream>>>(
      Cw, Wb + 3 * DD, bq, bk, bo, nullptr, nullptr, nullptr, out, 3);
}
